// GNNEncoder_1408749273540
// MI455X (gfx1250) — matrix operand order repaired and checked by reading, not compiled, not run on hardware
//
#include <hip/hip_runtime.h>
#include <hip/hip_bf16.h>

// ---------------- problem constants (match reference setup_inputs) ----------
#define N_NODES 20000
#define N_EDGES 320000
#define N_GRAPH 64
#define HDIM    280
#define HP      288          // padded row stride for all activation tensors
#define NFEAT   5
#define EFEAT   4
#define EAP     32           // padded edge_attr row stride
#define NLAYERS 3
#define ZDIM    64
#define KV_MSG  608          // virtual K: 288(h[dst]) + 288(h[src]) + 32(edge_attr)
#define KV_UPD  576          // virtual K: 288(h) + 288(agg)
#define KV_H    288          // virtual K: 288(dense H)
#define NPADW   320          // weight column padding (mult of 64 for 16x64 tiles)
#define BN_EPS  1e-5f
#define NT      4            // 16-col tiles per wave (16x64 C tile)

typedef __attribute__((ext_vector_type(16))) __bf16 v16bf;
typedef __attribute__((ext_vector_type(8)))  float  v8f;

__device__ __forceinline__ unsigned short f2bf(float f) {
    unsigned u = __builtin_bit_cast(unsigned, f);
    u += 0x7FFFu + ((u >> 16) & 1u);           // round-to-nearest-even
    return (unsigned short)(u >> 16);
}
__device__ __forceinline__ float bf2f(unsigned short h) {
    unsigned u = ((unsigned)h) << 16;
    return __builtin_bit_cast(float, u);
}

// ---------------- small elementwise kernels ---------------------------------
__global__ void k_input_layer(const float* __restrict__ x, const float* __restrict__ Wi,
                              const float* __restrict__ bi, float* __restrict__ h,
                              unsigned short* __restrict__ hbf) {
    int idx = blockIdx.x * blockDim.x + threadIdx.x;
    if (idx >= N_NODES * HP) return;
    int n = idx / HP, j = idx % HP;
    float s = 0.0f;
    if (j < HDIM) {
        s = bi[j];
#pragma unroll
        for (int k = 0; k < NFEAT; ++k) s += x[n * NFEAT + k] * Wi[k * HDIM + j];
    }
    h[idx] = s;
    hbf[idx] = f2bf(s);
}

__global__ void k_ea_prep(const float* __restrict__ ea, unsigned short* __restrict__ eabf) {
    int idx = blockIdx.x * blockDim.x + threadIdx.x;
    if (idx >= N_EDGES * EAP) return;
    int e = idx / EAP, k = idx % EAP;
    eabf[idx] = (k < EFEAT) ? f2bf(ea[e * EFEAT + k]) : (unsigned short)0;
}

__global__ void k_f32_to_bf16(const float* __restrict__ in, unsigned short* __restrict__ out, int n) {
    int idx = blockIdx.x * blockDim.x + threadIdx.x;
    if (idx < n) out[idx] = f2bf(in[idx]);
}

// W fp32 [Kreal x Nn] row-major -> Wt bf16 [NPADW x Kv] transposed, virtual-K padded.
__global__ void k_prep_weight(const float* __restrict__ W, unsigned short* __restrict__ Wt,
                              int Nn, int Kv, int s1v, int s2v,
                              int s1r, int s2r, int s3r) {
    int idx = blockIdx.x * blockDim.x + threadIdx.x;
    if (idx >= NPADW * Kv) return;
    int n = idx / Kv, k = idx % Kv;
    int rk = -1;
    if (k < s1v)            { if (k < s1r) rk = k; }
    else if (k < s1v + s2v) { int kk = k - s1v;       if (kk < s2r) rk = s1r + kk; }
    else                    { int kk = k - s1v - s2v; if (kk < s3r) rk = s1r + s2r + kk; }
    float v = (n < Nn && rk >= 0) ? W[rk * Nn + n] : 0.0f;
    Wt[idx] = f2bf(v);
}

// In-place BN (batch stats from colsum/colsum2) + ReLU over bf16 tensor [rows x HP]
__global__ void k_bn_relu(unsigned short* __restrict__ buf,
                          const float* __restrict__ colsum, const float* __restrict__ colsum2,
                          const float* __restrict__ g, const float* __restrict__ be,
                          long total, float invCnt) {
    long idx = (long)blockIdx.x * blockDim.x + threadIdx.x;
    if (idx >= total) return;
    int c = (int)(idx % HP);
    if (c >= HDIM) return;
    float m  = colsum[c] * invCnt;
    float v  = colsum2[c] * invCnt - m * m;
    float xv = bf2f(buf[idx]);
    float y  = g[c] * (xv - m) * __frsqrt_rn(v + BN_EPS) + be[c];
    buf[idx] = f2bf(fmaxf(y, 0.0f));
}

// scatter-add of messages into destination nodes (padded stride)
__global__ void k_aggregate(const unsigned short* __restrict__ msg, const int* __restrict__ dst,
                            float* __restrict__ agg) {
    long idx = (long)blockIdx.x * blockDim.x + threadIdx.x;
    if (idx >= (long)N_EDGES * HP) return;
    int e = (int)(idx / HP), c = (int)(idx % HP);
    if (c >= HDIM) return;
    atomicAdd(&agg[(long)dst[e] * HP + c], bf2f(msg[idx]));
}

// residual: h += upd ; refresh bf16 copy (pads stay 0 since upd pads are 0)
__global__ void k_update_h(float* __restrict__ h, unsigned short* __restrict__ hbf,
                           const unsigned short* __restrict__ upd) {
    int idx = blockIdx.x * blockDim.x + threadIdx.x;
    if (idx >= N_NODES * HP) return;
    float v = h[idx] + bf2f(upd[idx]);
    h[idx] = v;
    hbf[idx] = f2bf(v);
}

__global__ void k_pool_scatter(const float* __restrict__ h, const int* __restrict__ batch,
                               float* __restrict__ pool, float* __restrict__ cnt) {
    int idx = blockIdx.x * blockDim.x + threadIdx.x;
    if (idx >= N_NODES * HP) return;
    int n = idx / HP, c = idx % HP;
    if (c >= HDIM) return;
    int b = batch[n];
    atomicAdd(&pool[b * HDIM + c], h[idx]);
    if (c == 0) atomicAdd(&cnt[b], 1.0f);
}

__global__ void k_head(const float* __restrict__ pool, const float* __restrict__ cnt,
                       const float* __restrict__ Wmu, const float* __restrict__ bmu,
                       float* __restrict__ out) {
    int idx = blockIdx.x * blockDim.x + threadIdx.x;
    if (idx >= N_GRAPH * ZDIM) return;
    int gI = idx / ZDIM, z = idx % ZDIM;
    float inv = 1.0f / fmaxf(cnt[gI], 1.0f);
    float s = bmu[z];
    for (int c = 0; c < HDIM; ++c) s += pool[gI * HDIM + c] * inv * Wmu[c * ZDIM + z];
    out[idx] = s;
}

// ---------------- WMMA GEMM with fused bias + BN-stats epilogue --------------
// C[rows x HP] = A[rows x KV] * Wt^T + bias ; bf16 out ; column sum/sumsq stats.
// MODE/KV are compile-time: the K loop is split into per-segment loops with
// loop-invariant base pointers -> zero per-chunk control flow; A fragment is
// 2x b128, each of the NT B fragments is 2x b128, then wmma. 16x64 C / wave.
//   MODE 0: A dense bf16 [rows x HP]                 (KV = 288)
//   MODE 1: A = [h[dst] | h[src] | edge_attr_pad]    (KV = 608)
//   MODE 2: A = [h | agg]                            (KV = 576)
template <int MODE, int KV>
__global__ void k_gemm_wmma(const unsigned short* __restrict__ hbf,
                            const unsigned short* __restrict__ eabf,
                            const unsigned short* __restrict__ aggbf,
                            const int* __restrict__ srcIdx, const int* __restrict__ dstIdx,
                            const unsigned short* __restrict__ denseA,
                            const unsigned short* __restrict__ Wt,
                            const float* __restrict__ bias,
                            unsigned short* __restrict__ outBf,
                            float* __restrict__ colsum, float* __restrict__ colsum2,
                            int rowTiles) {
    const int lane = threadIdx.x & 31;
    const int wid  = threadIdx.x >> 5;
    const int tm   = blockIdx.x * 4 + wid;
    if (tm >= rowTiles) return;                   // uniform whole-wave exit
    const int hi   = lane >> 4;                   // lane half: 0 | 1
    const int l16  = lane & 15;
    const int arow = tm * 16 + l16;               // A row this lane feeds
    const int cg   = blockIdx.y * (NT * 16);      // first column of this group

    // weight-column pointers, hoisted out of the K loops (advance by kofs only)
    const unsigned short* wb[NT];
#pragma unroll
    for (int t = 0; t < NT; ++t)
        wb[t] = Wt + (long)(cg + t * 16 + l16) * KV + hi * 8;

    v8f acc[NT] = {};

    auto run_seg = [&](const unsigned short* base, int nch, int kofs) {
        const unsigned short* ap = base + hi * 8;
        for (int c = 0; c < nch; ++c) {
            union { uint4 q[2]; v16bf v; } a;
            a.q[0] = *(const uint4*)(ap + c * 32);        // K = kc + hi*8 + e
            a.q[1] = *(const uint4*)(ap + c * 32 + 16);   // K = kc + hi*8 + 16 + e
#pragma unroll
            for (int t = 0; t < NT; ++t) {
                const unsigned short* w = wb[t] + kofs + c * 32;
                union { uint4 q[2]; v16bf v; } b;
                b.q[0] = *(const uint4*)(w);
                b.q[1] = *(const uint4*)(w + 16);
                acc[t] = __builtin_amdgcn_wmma_f32_16x16x32_bf16(
                    false, a.v, false, b.v, (short)0, acc[t], false, false);
            }
        }
    };

    if (MODE == 1) {
        const unsigned short* ab0 = hbf + (long)dstIdx[arow] * HP;
        const unsigned short* ab1 = hbf + (long)srcIdx[arow] * HP;
        const unsigned short* ab2 = eabf + (long)arow * EAP;
        run_seg(ab0, HP / 32, 0);           // 9 chunks: h[dst]
        run_seg(ab1, HP / 32, HP);          // 9 chunks: h[src]
        run_seg(ab2, EAP / 32, 2 * HP);     // 1 chunk : edge_attr
    } else if (MODE == 2) {
        run_seg(hbf + (long)arow * HP, HP / 32, 0);     // 9 chunks: h
        run_seg(aggbf + (long)arow * HP, HP / 32, HP);  // 9 chunks: agg
    } else {
        run_seg(denseA + (long)arow * HP, HP / 32, 0);  // 9 chunks: dense
    }

#pragma unroll
    for (int t = 0; t < NT; ++t) {
        const int col = cg + t * 16 + l16;
        const bool cok = (col < HDIM);
        const float bv = cok ? bias[col] : 0.0f;
        float s = 0.0f, s2 = 0.0f;
#pragma unroll
        for (int r = 0; r < 8; ++r) {
            const int m = tm * 16 + r + hi * 8;   // ISA C-layout: VGPR r + lane half
            const float y = acc[t][r] + bv;
            if (col < HP) outBf[(long)m * HP + col] = cok ? f2bf(y) : (unsigned short)0;
            if (cok) { s += y; s2 += y * y; }
        }
        if (cok) { atomicAdd(&colsum[col], s); atomicAdd(&colsum2[col], s2); }
    }
}

// ---------------- host side --------------------------------------------------
static inline char* bump(char*& p, size_t bytes) {
    char* r = p;
    p += (bytes + 255) & ~(size_t)255;
    return r;
}
static inline int blk(long n) { return (int)((n + 255) / 256); }

extern "C" void kernel_launch(void* const* d_in, const int* in_sizes, int n_in,
                              void* d_out, int out_size, void* d_ws, size_t ws_size,
                              hipStream_t stream) {
    const float* x   = (const float*)d_in[0];
    const float* ea  = (const float*)d_in[1];
    const int* eidx  = (const int*)d_in[2];      // [2,E]: row0 = src, row1 = dst
    const int* batch = (const int*)d_in[3];
    const float* Wi  = (const float*)d_in[4];
    const float* bi  = (const float*)d_in[5];
    const float* Wm1 = (const float*)d_in[6];
    const float* bm1 = (const float*)d_in[7];
    const float* g1  = (const float*)d_in[8];
    const float* be1 = (const float*)d_in[9];
    const float* Wm2 = (const float*)d_in[10];
    const float* bm2 = (const float*)d_in[11];
    const float* g2  = (const float*)d_in[12];
    const float* be2 = (const float*)d_in[13];
    const float* Wu1 = (const float*)d_in[14];
    const float* bu1 = (const float*)d_in[15];
    const float* g3  = (const float*)d_in[16];
    const float* be3 = (const float*)d_in[17];
    const float* Wu2 = (const float*)d_in[18];
    const float* bu2 = (const float*)d_in[19];
    const float* g4  = (const float*)d_in[20];
    const float* be4 = (const float*)d_in[21];
    const float* Wmu = (const float*)d_in[22];
    const float* bmu = (const float*)d_in[23];
    const int* srcIdx = eidx;
    const int* dstIdx = eidx + N_EDGES;

    // ---- workspace carve ----
    char* p = (char*)d_ws;
    float*          h     = (float*)         bump(p, (size_t)N_NODES * HP * 4);
    unsigned short* hbf   = (unsigned short*)bump(p, (size_t)N_NODES * HP * 2);
    unsigned short* eabf  = (unsigned short*)bump(p, (size_t)N_EDGES * EAP * 2);
    float*          aggf  = (float*)         bump(p, (size_t)N_NODES * HP * 4);
    unsigned short* aggbf = (unsigned short*)bump(p, (size_t)N_NODES * HP * 2);
    unsigned short* bufA  = (unsigned short*)bump(p, (size_t)N_EDGES * HP * 2);
    unsigned short* bufB  = (unsigned short*)bump(p, (size_t)N_EDGES * HP * 2);
    unsigned short* WtBuf = (unsigned short*)bump(p, (size_t)NPADW * KV_MSG * 2);
    float*          csum  = (float*)         bump(p, (size_t)NPADW * 4);
    float*          csq   = (float*)         bump(p, (size_t)NPADW * 4);
    float*          pool  = (float*)         bump(p, (size_t)N_GRAPH * HDIM * 4);
    float*          cnt   = (float*)         bump(p, (size_t)N_GRAPH * 4);
    (void)ws_size; (void)n_in; (void)in_sizes; (void)out_size;

    const int rtE = N_EDGES / 16;   // 20000 edge row-tiles
    const int rtN = N_NODES / 16;   // 1250  node row-tiles
    const dim3 gridE((rtE + 3) / 4, NPADW / (NT * 16));   // y = 5 col groups
    const dim3 gridN((rtN + 3) / 4, NPADW / (NT * 16));

    // ---- preamble ----
    k_input_layer<<<blk((long)N_NODES * HP), 256, 0, stream>>>(x, Wi, bi, h, hbf);
    k_ea_prep<<<blk((long)N_EDGES * EAP), 256, 0, stream>>>(ea, eabf);

    for (int l = 0; l < NLAYERS; ++l) {
        // ---- message MLP layer 1: [E,564] x [564,280] (virtual K = 608) ----
        k_prep_weight<<<blk((long)NPADW * KV_MSG), 256, 0, stream>>>(
            Wm1 + (size_t)l * (2 * HDIM + EFEAT) * HDIM, WtBuf,
            HDIM, KV_MSG, HP, HP, HDIM, HDIM, EFEAT);
        hipMemsetAsync(csum, 0, NPADW * 4, stream);
        hipMemsetAsync(csq,  0, NPADW * 4, stream);
        k_gemm_wmma<1, KV_MSG><<<gridE, 128, 0, stream>>>(hbf, eabf, nullptr, srcIdx, dstIdx,
            nullptr, WtBuf, bm1 + l * HDIM, bufA, csum, csq, rtE);
        k_bn_relu<<<blk((long)N_EDGES * HP), 256, 0, stream>>>(
            bufA, csum, csq, g1 + l * HDIM, be1 + l * HDIM, (long)N_EDGES * HP, 1.0f / N_EDGES);

        // ---- message MLP layer 2: [E,280] x [280,280] (virtual K = 288) ----
        k_prep_weight<<<blk((long)NPADW * KV_H), 256, 0, stream>>>(
            Wm2 + (size_t)l * HDIM * HDIM, WtBuf, HDIM, KV_H, HP, 0, HDIM, 0, 0);
        hipMemsetAsync(csum, 0, NPADW * 4, stream);
        hipMemsetAsync(csq,  0, NPADW * 4, stream);
        k_gemm_wmma<0, KV_H><<<gridE, 128, 0, stream>>>(nullptr, nullptr, nullptr, nullptr, nullptr,
            bufA, WtBuf, bm2 + l * HDIM, bufB, csum, csq, rtE);
        k_bn_relu<<<blk((long)N_EDGES * HP), 256, 0, stream>>>(
            bufB, csum, csq, g2 + l * HDIM, be2 + l * HDIM, (long)N_EDGES * HP, 1.0f / N_EDGES);

        // ---- aggregate (scatter-add into dst) ----
        hipMemsetAsync(aggf, 0, (size_t)N_NODES * HP * 4, stream);
        k_aggregate<<<blk((long)N_EDGES * HP), 256, 0, stream>>>(bufB, dstIdx, aggf);
        k_f32_to_bf16<<<blk((long)N_NODES * HP), 256, 0, stream>>>(aggf, aggbf, N_NODES * HP);

        // ---- update MLP layer 1: [N,560] x [560,280] (virtual K = 576) ----
        k_prep_weight<<<blk((long)NPADW * KV_UPD), 256, 0, stream>>>(
            Wu1 + (size_t)l * (2 * HDIM) * HDIM, WtBuf,
            HDIM, KV_UPD, HP, HP, HDIM, HDIM, 0);
        hipMemsetAsync(csum, 0, NPADW * 4, stream);
        hipMemsetAsync(csq,  0, NPADW * 4, stream);
        k_gemm_wmma<2, KV_UPD><<<gridN, 128, 0, stream>>>(hbf, nullptr, aggbf, nullptr, nullptr,
            nullptr, WtBuf, bu1 + l * HDIM, bufA, csum, csq, rtN);
        k_bn_relu<<<blk((long)N_NODES * HP), 256, 0, stream>>>(
            bufA, csum, csq, g3 + l * HDIM, be3 + l * HDIM, (long)N_NODES * HP, 1.0f / N_NODES);

        // ---- update MLP layer 2: [N,280] x [280,280] (virtual K = 288) ----
        k_prep_weight<<<blk((long)NPADW * KV_H), 256, 0, stream>>>(
            Wu2 + (size_t)l * HDIM * HDIM, WtBuf, HDIM, KV_H, HP, 0, HDIM, 0, 0);
        hipMemsetAsync(csum, 0, NPADW * 4, stream);
        hipMemsetAsync(csq,  0, NPADW * 4, stream);
        k_gemm_wmma<0, KV_H><<<gridN, 128, 0, stream>>>(nullptr, nullptr, nullptr, nullptr, nullptr,
            bufA, WtBuf, bu2 + l * HDIM, bufB, csum, csq, rtN);
        k_bn_relu<<<blk((long)N_NODES * HP), 256, 0, stream>>>(
            bufB, csum, csq, g4 + l * HDIM, be4 + l * HDIM, (long)N_NODES * HP, 1.0f / N_NODES);

        // ---- residual ----
        k_update_h<<<blk((long)N_NODES * HP), 256, 0, stream>>>(h, hbf, bufB);
    }

    // ---- global mean pool + head ----
    hipMemsetAsync(pool, 0, (size_t)N_GRAPH * HDIM * 4, stream);
    hipMemsetAsync(cnt,  0, (size_t)N_GRAPH * 4, stream);
    k_pool_scatter<<<blk((long)N_NODES * HP), 256, 0, stream>>>(h, batch, pool, cnt);
    k_head<<<blk((long)N_GRAPH * ZDIM), 256, 0, stream>>>(pool, cnt, Wmu, bmu, (float*)d_out);
}
